// gwnet_59811714564163
// MI455X (gfx1250) — matrix operand order repaired and checked by reading, not compiled, not run on hardware
//
#include <hip/hip_runtime.h>
#include <hip/hip_bf16.h>
#include <cstdint>

typedef __bf16 bf16;
typedef __attribute__((ext_vector_type(4)))  __bf16 v4bf;
typedef __attribute__((ext_vector_type(8)))  __bf16 v8bf;
typedef __attribute__((ext_vector_type(16))) __bf16 v16bf;
typedef __attribute__((ext_vector_type(8)))  float  v8f;
typedef __attribute__((ext_vector_type(2)))  int    v2i;

#define ASYNCOK (__has_builtin(__builtin_amdgcn_global_load_async_to_lds_b64) && \
                 __has_builtin(__builtin_amdgcn_s_wait_asynccnt))

__device__ __forceinline__ v8f wmma_bf16(v16bf a, v16bf b, v8f c) {
  return __builtin_amdgcn_wmma_f32_16x16x32_bf16(false, a, false, b, (short)0, c, false, false);
}

// A fragment from a bf16 row: per lane one contiguous 16-elem run, k = 16*half + s
__device__ __forceinline__ v16bf afrag_bf(const bf16* row, int half) {
  v8bf lo = *(const v8bf*)(row + half * 16);
  v8bf hi = *(const v8bf*)(row + half * 16 + 8);
  return __builtin_shufflevector(lo, hi, 0,1,2,3,4,5,6,7,8,9,10,11,12,13,14,15);
}

#if ASYNCOK
typedef __attribute__((address_space(1))) v2i g_v2i;  // global (AS1) b64 payload
typedef __attribute__((address_space(3))) v2i l_v2i;  // LDS (AS3) b64 payload
__device__ __forceinline__ void async_copy_b64(const void* g, void* l) {
  __builtin_amdgcn_global_load_async_to_lds_b64(
      (g_v2i*)(unsigned long long)(uintptr_t)g,
      (l_v2i*)(unsigned int)(uintptr_t)l, 0, 0);
}
#endif

// ---------------------------------------------------------------------------
// Prep: transposed supports + fragment-order packed bf16 weights
// packed flat index = ((kc*numOt + ot)*32 + lane)*16 + j ;
//   k = kc*32 + (lane>>4)*16 + j , o = ot*16 + (lane&15)
// ---------------------------------------------------------------------------
__global__ void prep_supT(const float* __restrict__ sup, bf16* __restrict__ supT) {
  int idx = blockIdx.x * blockDim.x + threadIdx.x;
  if (idx >= 2 * 1024 * 1024) return;
  int s = idx >> 20, r = idx & 1048575, n = r >> 10, m = r & 1023;
  supT[((size_t)s << 20) + ((size_t)m << 10) + n] = (bf16)sup[idx];
}

__global__ void adp_kernel(const float* __restrict__ nv1, const float* __restrict__ nv2,
                           bf16* __restrict__ adpT) {
  int n = blockIdx.x, tid = threadIdx.x;
  __shared__ float v1[10];
  __shared__ float red[256];
  if (tid < 10) v1[tid] = nv1[n * 10 + tid];
  __syncthreads();
  float vals[4];
  float mx = -1e30f;
  for (int j = 0; j < 4; ++j) {
    int m = tid + j * 256;
    float s = 0.f;
#pragma unroll
    for (int k = 0; k < 10; ++k) s += v1[k] * nv2[k * 1024 + m];
    s = s > 0.f ? s : 0.f;
    vals[j] = s; mx = fmaxf(mx, s);
  }
  red[tid] = mx; __syncthreads();
  for (int st = 128; st > 0; st >>= 1) { if (tid < st) red[tid] = fmaxf(red[tid], red[tid + st]); __syncthreads(); }
  mx = red[0]; __syncthreads();
  float sum = 0.f;
  for (int j = 0; j < 4; ++j) { vals[j] = __expf(vals[j] - mx); sum += vals[j]; }
  red[tid] = sum; __syncthreads();
  for (int st = 128; st > 0; st >>= 1) { if (tid < st) red[tid] += red[tid + st]; __syncthreads(); }
  float inv = 1.f / red[0];
  for (int j = 0; j < 4; ++j) {
    int m = tid + j * 256;
    adpT[((size_t)m << 10) + n] = (bf16)(vals[j] * inv);
  }
}

__global__ void prep_dconv_w(const float* __restrict__ fw, const float* __restrict__ gw,
                             bf16* __restrict__ fo, bf16* __restrict__ go) {
  int idx = blockIdx.x * blockDim.x + threadIdx.x;
  if (idx >= 8 * 2048) return;
  int i = idx >> 11, r = idx & 2047;
  int j = r & 15, lane = (r >> 4) & 31, kt_ot = r >> 9;
  int kt = kt_ot >> 1, ot = kt_ot & 1;
  int k = kt * 32 + (lane >> 4) * 16 + j;
  int o = ot * 16 + (lane & 15);
  int tap = k >> 5, c = k & 31;
  size_t si = (((size_t)i * 32 + o) * 32 + c) * 2 + tap;
  fo[idx] = (bf16)fw[si];
  go[idx] = (bf16)gw[si];
}

__global__ void prep_skip_w(const float* __restrict__ sw, bf16* __restrict__ dst) {
  int idx = blockIdx.x * blockDim.x + threadIdx.x;
  if (idx >= 8 * 8192) return;
  int i = idx >> 13, r = idx & 8191;
  int j = r & 15, lane = (r >> 4) & 31, ot = r >> 9;
  int k = (lane >> 4) * 16 + j;
  int o = ot * 16 + (lane & 15);
  dst[idx] = (bf16)sw[((size_t)i * 256 + o) * 32 + k];
}

__global__ void prep_gcn_w(const float* __restrict__ gwm, bf16* __restrict__ dst) {
  int idx = blockIdx.x * blockDim.x + threadIdx.x;
  if (idx >= 8 * 7168) return;
  int i = idx / 7168, r = idx - i * 7168;
  int j = r & 15, lane = (r >> 4) & 31, blk_ot = r >> 9;
  int jb = blk_ot >> 1, ot = blk_ot & 1;
  int k = jb * 32 + (lane >> 4) * 16 + j;
  int o = ot * 16 + (lane & 15);
  dst[idx] = (bf16)gwm[((size_t)i * 32 + o) * 224 + k];
}

__global__ void prep_end_w(const float* __restrict__ e1, const float* __restrict__ e2,
                           bf16* __restrict__ o1, bf16* __restrict__ o2) {
  int idx = blockIdx.x * blockDim.x + threadIdx.x;
  if (idx < 131072) {
    int j = idx & 15, lane = (idx >> 4) & 31, kc_ot = idx >> 9;
    int kc = kc_ot >> 5, ot = kc_ot & 31;
    int k = kc * 32 + (lane >> 4) * 16 + j;
    int o = ot * 16 + (lane & 15);
    o1[idx] = (bf16)e1[(size_t)o * 256 + k];
  }
  if (idx < 8192) {
    int j = idx & 15, lane = (idx >> 4) & 31, kc = idx >> 9;
    int k = kc * 32 + (lane >> 4) * 16 + j;
    int o = lane & 15;
    o2[idx] = (o < 12) ? (bf16)e2[(size_t)o * 512 + k] : (bf16)0.0f;
  }
}

__global__ void zero_f32(float* __restrict__ p, int n) {
  int i = blockIdx.x * blockDim.x + threadIdx.x;
  if (i < n) p[i] = 0.f;
}

__global__ void start_conv(const float* __restrict__ xin, const float* __restrict__ w,
                           const float* __restrict__ b, float* __restrict__ xout) {
  int row = blockIdx.x * blockDim.x + threadIdx.x;
  if (row >= 16 * 13 * 1024) return;
  int n = row & 1023, bt = row >> 10, bb = bt / 13, t = bt - bb * 13;
  float v0 = xin[((size_t)(bb * 2 + 0) * 1024 + n) * 13 + t];
  float v1 = xin[((size_t)(bb * 2 + 1) * 1024 + n) * 13 + t];
  float* o = xout + (size_t)row * 32;
#pragma unroll
  for (int c = 0; c < 32; ++c) o[c] = w[c * 2] * v0 + w[c * 2 + 1] * v1 + b[c];
}

// ---------------------------------------------------------------------------
// Gated TCN -> bf16 activations. One wave: 16 rows x 32 outs, filter+gate, 8 WMMAs.
// ---------------------------------------------------------------------------
__global__ void dconv_wmma(const float* __restrict__ xin, const bf16* __restrict__ fw,
                           const bf16* __restrict__ gw, const float* __restrict__ fb,
                           const float* __restrict__ gb, bf16* __restrict__ xout,
                           int t_in, int t_out, int d) {
  int wave = (blockIdx.x * blockDim.x + threadIdx.x) >> 5;
  int lane = threadIdx.x & 31, half = lane >> 4, l16 = lane & 15;
  if (wave >= t_out * 1024) return;
  int m0 = wave << 4;
  int rowg = m0 + l16;
  int bt = rowg >> 10, n = rowg & 1023;
  int b = bt / t_out, tau = bt - b * t_out;
  const float* r0 = xin + ((size_t)(b * t_in + tau) * 1024 + n) * 32;
  const float* r1 = r0 + (size_t)d * 1024 * 32;
  v8f aF0 = {}, aF1 = {}, aG0 = {}, aG1 = {};
#pragma unroll
  for (int kt = 0; kt < 2; ++kt) {
    const float* base = (kt ? r1 : r0) + half * 16;
    v16bf a;
#pragma unroll
    for (int j = 0; j < 8; ++j) a[j] = (bf16)base[j];
#pragma unroll
    for (int j = 0; j < 8; ++j) a[8 + j] = (bf16)base[8 + j];
    v16bf bf0 = *(const v16bf*)(fw + ((kt * 2 + 0) * 32 + lane) * 16);
    v16bf bf1 = *(const v16bf*)(fw + ((kt * 2 + 1) * 32 + lane) * 16);
    v16bf bg0 = *(const v16bf*)(gw + ((kt * 2 + 0) * 32 + lane) * 16);
    v16bf bg1 = *(const v16bf*)(gw + ((kt * 2 + 1) * 32 + lane) * 16);
    aF0 = wmma_bf16(a, bf0, aF0);  aF1 = wmma_bf16(a, bf1, aF1);
    aG0 = wmma_bf16(a, bg0, aG0);  aG1 = wmma_bf16(a, bg1, aG1);
  }
#pragma unroll
  for (int v = 0; v < 8; ++v) {
    int m = m0 + v + half * 8;
    float f0 = tanhf(aF0[v] + fb[l16]);
    float g0 = 1.f / (1.f + __expf(-(aG0[v] + gb[l16])));
    float f1 = tanhf(aF1[v] + fb[16 + l16]);
    float g1 = 1.f / (1.f + __expf(-(aG1[v] + gb[16 + l16])));
    xout[(size_t)m * 32 + l16]      = (bf16)(f0 * g0);
    xout[(size_t)m * 32 + 16 + l16] = (bf16)(f1 * g1);
  }
}

// skip conv at the last surviving time slice
__global__ void skip_wmma(const bf16* __restrict__ xnew, const bf16* __restrict__ wt,
                          const float* __restrict__ sb, float* __restrict__ skip, int t_out) {
  int wave = (blockIdx.x * blockDim.x + threadIdx.x) >> 5;
  int lane = threadIdx.x & 31, half = lane >> 4, l16 = lane & 15;
  if (wave >= 16384) return;
  int m0 = (wave >> 4) << 4, ot = wave & 15, o0 = ot << 4;
  int rowg = m0 + l16;
  int b = rowg >> 10, n = rowg & 1023;
  const bf16* rp = xnew + ((size_t)(b * t_out + (t_out - 1)) * 1024 + n) * 32;
  v16bf a = afrag_bf(rp, half);
  v16bf bv = *(const v16bf*)(wt + (ot * 32 + lane) * 16);
  v8f acc = {};
  acc = wmma_bf16(a, bv, acc);
#pragma unroll
  for (int v = 0; v < 8; ++v) {
    int m = m0 + v + half * 8;
    size_t oi = (size_t)m * 256 + o0 + l16;
    skip[oi] += acc[v] + sb[o0 + l16];
  }
}

// ---------------------------------------------------------------------------
// Diffusion GEMM: per (b,t), dst = A^T(1024x1024 bf16) @ X(1024x32 bf16).
// Block: 8 waves, 256 rows, full 32 cols. X panel (32x32) staged per K-step:
// async-copied raw [k][c] (double buffered), then transposed into LDS [c][k]
// so B fragments are contiguous ds_b128 loads. Each wave: 2x2 tiles, 4 WMMA/iter.
// ---------------------------------------------------------------------------
__global__ void __launch_bounds__(256) diffuse_wmma(const bf16* __restrict__ src,
                                                    bf16* __restrict__ dst,
                                                    const bf16* __restrict__ AT,
                                                    int bt_count) {
  __shared__ __align__(16) bf16 rawbuf[2][1024]; // [k][c]
  __shared__ __align__(32) bf16 Tbuf[1024];      // [c][k]
  int tid = threadIdx.x;
  int wave = tid >> 5, lane = tid & 31, half = lane >> 4, l16 = lane & 15;
  int bt = blockIdx.x >> 2, mblk = blockIdx.x & 3;
  const bf16* X = src + (size_t)bt * 32768;
  int m0 = mblk * 256 + wave * 32;
  const bf16* arow0 = AT + (size_t)(m0 + l16) * 1024;
  const bf16* arow1 = arow0 + 16 * 1024;
  v8f a00 = {}, a01 = {}, a10 = {}, a11 = {};
  int cur = 0;
#if ASYNCOK
  async_copy_b64(X + tid * 4, &rawbuf[0][tid * 4]);
#endif
  for (int k0 = 0; k0 < 1024; k0 += 32) {
    v4bf q;
#if ASYNCOK
    __builtin_amdgcn_s_wait_asynccnt(0);
    if (k0 + 32 < 1024)
      async_copy_b64(X + (k0 + 32) * 32 + tid * 4, &rawbuf[cur ^ 1][tid * 4]);
    q = *(const v4bf*)(&rawbuf[cur][tid * 4]); // own slice: self-written, self-waited
#else
    q = *(const v4bf*)(X + k0 * 32 + tid * 4);
#endif
    int kk = tid >> 3, cb = (tid & 7) * 4;
#pragma unroll
    for (int e = 0; e < 4; ++e) Tbuf[(cb + e) * 32 + kk] = q[e];
    __syncthreads();
    v16bf af0 = afrag_bf(arow0 + k0, half);
    v16bf af1 = afrag_bf(arow1 + k0, half);
    v16bf b0 = *(const v16bf*)(&Tbuf[l16 * 32 + half * 16]);
    v16bf b1 = *(const v16bf*)(&Tbuf[(16 + l16) * 32 + half * 16]);
    a00 = wmma_bf16(af0, b0, a00);
    a01 = wmma_bf16(af0, b1, a01);
    a10 = wmma_bf16(af1, b0, a10);
    a11 = wmma_bf16(af1, b1, a11);
    __syncthreads();
    cur ^= 1;
  }
  bf16* D = dst + (size_t)bt * 32768;
#pragma unroll
  for (int v = 0; v < 8; ++v) {
    int mA = m0 + v + half * 8, mB = mA + 16;
    D[(size_t)mA * 32 + l16]      = (bf16)a00[v];
    D[(size_t)mA * 32 + 16 + l16] = (bf16)a01[v];
    D[(size_t)mB * 32 + l16]      = (bf16)a10[v];
    D[(size_t)mB * 32 + 16 + l16] = (bf16)a11[v];
  }
}

// gcn accumulation of one 32x32 weight block (K=224 conv split into 7 blocks)
__global__ void gcn_wmma(const bf16* __restrict__ src, const bf16* __restrict__ wt,
                         float* __restrict__ out, const float* __restrict__ resid,
                         const float* __restrict__ bias, int t_in, int t_out, int mode) {
  int wave = (blockIdx.x * blockDim.x + threadIdx.x) >> 5;
  int lane = threadIdx.x & 31, half = lane >> 4, l16 = lane & 15;
  if (wave >= t_out * 1024 * 2) return;
  int m0 = (wave >> 1) << 4, ot = wave & 1, o0 = ot << 4;
  int rowg = m0 + l16;
  v16bf a = afrag_bf(src + (size_t)rowg * 32, half);
  v16bf bv = *(const v16bf*)(wt + (ot * 32 + lane) * 16);
  v8f acc = {};
  acc = wmma_bf16(a, bv, acc);
#pragma unroll
  for (int v = 0; v < 8; ++v) {
    int m = m0 + v + half * 8;
    size_t oi = (size_t)m * 32 + o0 + l16;
    if (mode == 0) {
      int bt = m >> 10, n = m & 1023;
      int b = bt / t_out, tau = bt - b * t_out;
      size_t ri = ((size_t)(b * t_in + tau + (t_in - t_out)) * 1024 + n) * 32 + o0 + l16;
      out[oi] = acc[v] + resid[ri] + bias[o0 + l16];
    } else {
      out[oi] += acc[v];
    }
  }
}

__global__ void bn_stats(const float* __restrict__ x, int M, float* __restrict__ stats) {
  int c = blockIdx.x, tid = threadIdx.x;
  float s = 0.f, s2 = 0.f;
  for (int r = tid; r < M; r += 256) { float v = x[(size_t)r * 32 + c]; s += v; s2 += v * v; }
  __shared__ float sh[256], sh2[256];
  sh[tid] = s; sh2[tid] = s2; __syncthreads();
  for (int st = 128; st > 0; st >>= 1) {
    if (tid < st) { sh[tid] += sh[tid + st]; sh2[tid] += sh2[tid + st]; }
    __syncthreads();
  }
  if (tid == 0) {
    float mu = sh[0] / (float)M;
    float var = sh2[0] / (float)M - mu * mu;
    stats[c] = mu; stats[32 + c] = rsqrtf(var + 1e-5f);
  }
}

__global__ void bn_apply(float* __restrict__ x, int total, const float* __restrict__ stats,
                         const float* __restrict__ g, const float* __restrict__ bb) {
  int i = blockIdx.x * blockDim.x + threadIdx.x;
  if (i >= total) return;
  int c = i & 31;
  x[i] = (x[i] - stats[c]) * stats[32 + c] * g[c] + bb[c];
}

__global__ void end1_wmma(const float* __restrict__ skip, const bf16* __restrict__ wt,
                          const float* __restrict__ b1, float* __restrict__ h1) {
  int wave = (blockIdx.x * blockDim.x + threadIdx.x) >> 5;
  int lane = threadIdx.x & 31, half = lane >> 4, l16 = lane & 15;
  if (wave >= 32768) return;
  int m0 = (wave >> 5) << 4, ot = wave & 31, o0 = ot << 4;
  const float* rp = skip + (size_t)(m0 + l16) * 256;
  v8f acc = {};
  for (int kc = 0; kc < 8; ++kc) {
    const float* base = rp + kc * 32 + half * 16;
    v16bf a;
#pragma unroll
    for (int j = 0; j < 8; ++j) a[j] = (bf16)fmaxf(base[j], 0.f);
#pragma unroll
    for (int j = 0; j < 8; ++j) a[8 + j] = (bf16)fmaxf(base[8 + j], 0.f);
    v16bf bv = *(const v16bf*)(wt + ((kc * 32 + ot) * 32 + lane) * 16);
    acc = wmma_bf16(a, bv, acc);
  }
#pragma unroll
  for (int v = 0; v < 8; ++v) {
    int m = m0 + v + half * 8;
    h1[(size_t)m * 512 + o0 + l16] = fmaxf(acc[v] + b1[o0 + l16], 0.f);
  }
}

__global__ void end2_wmma(const float* __restrict__ h1, const bf16* __restrict__ wt,
                          const float* __restrict__ b2, float* __restrict__ out) {
  int wave = (blockIdx.x * blockDim.x + threadIdx.x) >> 5;
  int lane = threadIdx.x & 31, half = lane >> 4, l16 = lane & 15;
  if (wave >= 1024) return;
  int m0 = wave << 4;
  const float* rp = h1 + (size_t)(m0 + l16) * 512;
  v8f acc = {};
  for (int kc = 0; kc < 16; ++kc) {
    const float* base = rp + kc * 32 + half * 16;
    v16bf a;
#pragma unroll
    for (int j = 0; j < 8; ++j) a[j] = (bf16)base[j];
#pragma unroll
    for (int j = 0; j < 8; ++j) a[8 + j] = (bf16)base[8 + j];
    v16bf bv = *(const v16bf*)(wt + (kc * 32 + lane) * 16);
    acc = wmma_bf16(a, bv, acc);
  }
  if (l16 < 12) {
#pragma unroll
    for (int v = 0; v < 8; ++v) {
      int m = m0 + v + half * 8;
      int b = m >> 10, n = m & 1023;
      out[((size_t)b * 12 + l16) * 1024 + n] = acc[v] + b2[l16];
    }
  }
}

// ---------------------------------------------------------------------------
extern "C" void kernel_launch(void* const* d_in, const int* in_sizes, int n_in,
                              void* d_out, int out_size, void* d_ws, size_t ws_size,
                              hipStream_t stream) {
  (void)in_sizes; (void)n_in; (void)out_size; (void)ws_size;
  const float* x_in     = (const float*)d_in[0];
  const float* sup      = (const float*)d_in[1];
  const float* nv1      = (const float*)d_in[2];
  const float* nv2      = (const float*)d_in[3];
  const float* start_w  = (const float*)d_in[4];
  const float* start_b  = (const float*)d_in[5];
  const float* filter_w = (const float*)d_in[6];
  const float* filter_b = (const float*)d_in[7];
  const float* gate_w   = (const float*)d_in[8];
  const float* gate_b   = (const float*)d_in[9];
  const float* skip_w   = (const float*)d_in[10];
  const float* skip_b   = (const float*)d_in[11];
  const float* gcn_w    = (const float*)d_in[12];
  const float* gcn_b    = (const float*)d_in[13];
  const float* bn_g     = (const float*)d_in[14];
  const float* bn_b     = (const float*)d_in[15];
  const float* end1_w   = (const float*)d_in[16];
  const float* end1_b   = (const float*)d_in[17];
  const float* end2_w   = (const float*)d_in[18];
  const float* end2_b   = (const float*)d_in[19];

  bf16* supT   = (bf16*)d_ws;                 // 3*1024*1024
  bf16* fwcat  = supT + 3145728;              // 8*2048 packed
  bf16* gwcat  = fwcat + 16384;
  bf16* skipwt = gwcat + 16384;               // 8*8192 packed
  bf16* gcnwt  = skipwt + 65536;              // 8*7*1024 packed
  bf16* e1wt   = gcnwt + 57344;               // 131072 packed
  bf16* e2wt   = e1wt + 131072;               // 8192 packed
  bf16* xnbf   = e2wt + 8192;                 // [16,12,1024,32] bf16
  bf16* x1bf   = xnbf + 6291456;
  bf16* x2bf   = x1bf + 6291456;
  float* xA    = (float*)(x2bf + 6291456);    // [16,13,1024,32] f32
  float* xB    = xA + 6815744;
  float* skipb = xB + 6815744;                // [16,1024,256]
  float* h1    = skipb + 4194304;             // [16,1024,512]
  float* stats = h1 + 8388608;                // 64

  auto cdiv = [](int a, int b) { return (a + b - 1) / b; };

  prep_supT   <<<cdiv(2097152, 256), 256, 0, stream>>>(sup, supT);
  adp_kernel  <<<1024, 256, 0, stream>>>(nv1, nv2, supT + 2 * 1048576);
  prep_dconv_w<<<cdiv(16384, 256), 256, 0, stream>>>(filter_w, gate_w, fwcat, gwcat);
  prep_skip_w <<<cdiv(65536, 256), 256, 0, stream>>>(skip_w, skipwt);
  prep_gcn_w  <<<cdiv(57344, 256), 256, 0, stream>>>(gcn_w, gcnwt);
  prep_end_w  <<<cdiv(131072, 256), 256, 0, stream>>>(end1_w, end2_w, e1wt, e2wt);
  zero_f32    <<<cdiv(4194304, 256), 256, 0, stream>>>(skipb, 4194304);
  start_conv  <<<cdiv(212992, 256), 256, 0, stream>>>(x_in, start_w, start_b, xA);

  const int tin[8] = {13, 12, 10, 9, 7, 6, 4, 3};
  const int dil[8] = {1, 2, 1, 2, 1, 2, 1, 2};
  float* xcur = xA;
  float* xpost = xB;

  for (int i = 0; i < 8; ++i) {
    int ti = tin[i], d = dil[i], to = ti - d;
    int bt = 16 * to;
    int M = bt * 1024;

    dconv_wmma<<<to * 256, 128, 0, stream>>>(
        xcur, fwcat + i * 2048, gwcat + i * 2048,
        filter_b + i * 32, gate_b + i * 32, xnbf, ti, to, d);

    skip_wmma<<<4096, 128, 0, stream>>>(xnbf, skipwt + i * 8192, skip_b + i * 256, skipb, to);

    gcn_wmma<<<to * 512, 128, 0, stream>>>(
        xnbf, gcnwt + (size_t)i * 7168, xpost, xcur, gcn_b + i * 32, ti, to, 0);

    for (int s = 0; s < 3; ++s) {
      const bf16* AT = supT + (size_t)s * 1048576;
      diffuse_wmma<<<bt * 4, 256, 0, stream>>>(xnbf, x1bf, AT, bt);
      gcn_wmma<<<to * 512, 128, 0, stream>>>(
          x1bf, gcnwt + (size_t)(i * 7 + 1 + 2 * s) * 1024, xpost, nullptr, nullptr, ti, to, 1);
      diffuse_wmma<<<bt * 4, 256, 0, stream>>>(x1bf, x2bf, AT, bt);
      gcn_wmma<<<to * 512, 128, 0, stream>>>(
          x2bf, gcnwt + (size_t)(i * 7 + 2 + 2 * s) * 1024, xpost, nullptr, nullptr, ti, to, 1);
    }

    bn_stats<<<32, 256, 0, stream>>>(xpost, M, stats);
    bn_apply<<<cdiv(M * 32, 256), 256, 0, stream>>>(xpost, M * 32, stats, bn_g + i * 32, bn_b + i * 32);

    float* tmp = xcur; xcur = xpost; xpost = tmp;
  }

  end1_wmma<<<8192, 128, 0, stream>>>(skipb, e1wt, end1_b, h1);
  end2_wmma<<<256, 128, 0, stream>>>(h1, e2wt, end2_b, (float*)d_out);
}
